// CriticNetwork_9929964389147
// MI455X (gfx1250) — compile-verified, matrix operand order checked by reading, not run on hardware
//
#include <hip/hip_runtime.h>
#include <hip/hip_bf16.h>

// ---------------------------------------------------------------------------
// CDNA5 (gfx1250) GCN critic.
//  - GEMMs: v_wmma_f32_16x16x32_bf16, 4 accumulators/wave, pre-packed B
//    fragments, A tile staged to LDS via global_load_async_to_lds_b128.
//  - GCN scatter: needed-mask filtered wave-per-edge f32 atomics.
// ---------------------------------------------------------------------------

typedef __attribute__((ext_vector_type(16))) __bf16 v16bf;
typedef __attribute__((ext_vector_type(8)))  float  v8f;

#define GN_NODES 50000
#define GD_IN    128
#define GD_HID   256
#define GFC1     512
#define GFC2     256
#define GN_ACT   64
#define GN_AGENTS 8192
#define G_LN_EPS 1e-5f

// ---------------------------------------------------------------------------
// B-matrix pre-pack:  B[K,N] fp32 (row-major)  ->  bf16 fragments laid out as
// P[kt][nt][lane][e] so a wave can load its 32x16 B fragment as one
// contiguous 32-byte vector per lane.
// Fragment map (16-bit operands, ISA 7.12.2):
//   element e, lane l :  K = kt*32 + (e&7) + (e>=8 ? 16 : 0) + ((l>=16)?8:0)
//                        N = nt*16 + (l&15)
// ---------------------------------------------------------------------------
__global__ void pack_b(const float* __restrict__ B, __bf16* __restrict__ P,
                       int N, int totTiles)
{
    int tile = blockIdx.x * (blockDim.x >> 5) + (threadIdx.x >> 5);
    if (tile >= totTiles) return;
    int ntiles = N >> 4;
    int kt = tile / ntiles;
    int nt = tile % ntiles;
    int lane = threadIdx.x & 31;
    int col  = nt * 16 + (lane & 15);
    int koff = (lane >> 4) * 8;
    const float* bp = B + (size_t)(kt * 32 + koff) * N + col;
    __bf16* pp = P + ((size_t)tile * 32 + lane) * 16;
#pragma unroll
    for (int e = 0; e < 8; ++e) {
        pp[e]     = (__bf16)bp[(size_t)e * N];
        pp[e + 8] = (__bf16)bp[(size_t)(e + 16) * N];
    }
}

// ---------------------------------------------------------------------------
// WMMA GEMM: C[M,N] = A[M,K] @ B[K,N] (+bias) (+relu)
// Block = 128 threads (4 waves). Block computes a 16 x 256 strip:
// each wave owns 4 adjacent 16x16 output tiles (4 accumulators, shared `a`).
// Per K-step the 16x32 fp32 A tile is staged once into LDS with
// global_load_async_to_lds_b128 (ASYNCcnt) and shared by all 4 waves.
// Requires M%16==0, N%256==0, K%32==0 (all call sites comply).
// ---------------------------------------------------------------------------
__global__ void gemm_bf16_wmma(const float* __restrict__ A,
                               const __bf16* __restrict__ BP,
                               const float* __restrict__ bias,
                               float* __restrict__ C,
                               int K, int ntiles /*N/16*/, int lda, int ldc,
                               int relu)
{
    const int tid  = threadIdx.x;
    const int lane = tid & 31;
    const int wave = tid >> 5;
    const int row0 = blockIdx.x * 16;
    const int ct0  = blockIdx.y * 16 + wave * 4;   // first of this wave's 4 col tiles
    const int l15  = lane & 15;
    const int koff = (lane >> 4) * 8;

    __shared__ float atile[16][32];                // 2 KB staged A tile
    const unsigned my_lds =
        (unsigned)(uintptr_t)(&atile[0][0]) +
        (unsigned)(((tid >> 3) * 32 + (tid & 7) * 4) * 4);   // row=tid/8, 16B chunk
    const float* agbase = A + (size_t)(row0 + (tid >> 3)) * lda + (tid & 7) * 4;

    v8f acc0 = {}, acc1 = {}, acc2 = {}, acc3 = {};

    const int kt_count = K >> 5;
    for (int kt = 0; kt < kt_count; ++kt) {
        __syncthreads();                           // previous reads done
        unsigned long long gp = (unsigned long long)(uintptr_t)(agbase + kt * 32);
        asm volatile("global_load_async_to_lds_b128 %0, %1, off"
                     :: "v"(my_lds), "v"(gp) : "memory");
        asm volatile("s_wait_asynccnt 0x0" ::: "memory");
        __syncthreads();                           // all waves' stages visible

        v16bf a;
#pragma unroll
        for (int e = 0; e < 8; ++e) {
            a[e]     = (__bf16)atile[l15][koff + e];
            a[e + 8] = (__bf16)atile[l15][koff + 16 + e];
        }
        const v16bf* bp = (const v16bf*)BP + ((size_t)kt * ntiles + ct0) * 32 + lane;
        v16bf b0 = bp[0];
        v16bf b1 = bp[32];
        v16bf b2 = bp[64];
        v16bf b3 = bp[96];
        acc0 = __builtin_amdgcn_wmma_f32_16x16x32_bf16(false, a, false, b0, (short)0, acc0, false, false);
        acc1 = __builtin_amdgcn_wmma_f32_16x16x32_bf16(false, a, false, b1, (short)0, acc1, false, false);
        acc2 = __builtin_amdgcn_wmma_f32_16x16x32_bf16(false, a, false, b2, (short)0, acc2, false, false);
        acc3 = __builtin_amdgcn_wmma_f32_16x16x32_bf16(false, a, false, b3, (short)0, acc3, false, false);
    }

    // C/D layout: row = row0 + r + ((lane>=16)?8:0), col = tile*16 + (lane&15)
    const int rb = row0 + (lane >> 4) * 8;
    const v8f accs[4] = {acc0, acc1, acc2, acc3};
#pragma unroll
    for (int j = 0; j < 4; ++j) {
        int   col = (ct0 + j) * 16 + l15;
        float bv  = bias ? bias[col] : 0.0f;
#pragma unroll
        for (int r = 0; r < 8; ++r) {
            float v = accs[j][r] + bv;
            if (relu) v = fmaxf(v, 0.0f);
            C[(size_t)(rb + r) * ldc + col] = v;
        }
    }
}

// ---------------------------------------------------------------------------
// GCN support kernels
// ---------------------------------------------------------------------------
__global__ void gcn_init(float* __restrict__ deg, int* __restrict__ needed, int n)
{
    int i = blockIdx.x * blockDim.x + threadIdx.x;
    if (i < n) { deg[i] = 1.0f; needed[i] = 0; }   // 1.0 accounts for self-loop
}

__global__ void gcn_degree(const int* __restrict__ dst, float* __restrict__ deg, int e)
{
    int i = blockIdx.x * blockDim.x + threadIdx.x;
    if (i < e) atomicAdd(&deg[dst[i]], 1.0f);
}

__global__ void gcn_dinv(float* __restrict__ deg, int n)
{
    int i = blockIdx.x * blockDim.x + threadIdx.x;
    if (i < n) {
        float d = deg[i];
        deg[i] = (d > 0.0f) ? rsqrtf(d) : 0.0f;
    }
}

__global__ void mark_needed(const int* __restrict__ idx, int* __restrict__ needed, int a)
{
    int i = blockIdx.x * blockDim.x + threadIdx.x;
    if (i < a) needed[idx[i]] = 1;
}

// h[i,:] = xw[i,:] * dinv[i]^2 + bg   (self-loop term + bias), needed rows only
__global__ void gcn_selfloop(const float* __restrict__ xw,
                             const float* __restrict__ dinv,
                             const int*   __restrict__ needed,
                             const float* __restrict__ bg,
                             float* __restrict__ h)
{
    int i = blockIdx.x;
    if (!needed[i]) return;
    int c = threadIdx.x;
    float di = dinv[i];
    h[(size_t)i * GD_HID + c] = xw[(size_t)i * GD_HID + c] * di * di + bg[c];
}

// wave-per-edge message scatter; only edges whose dst is an agent node matter
__global__ void gcn_scatter(const int*   __restrict__ src,
                            const int*   __restrict__ dst,
                            const float* __restrict__ xw,
                            const float* __restrict__ dinv,
                            const int*   __restrict__ needed,
                            float* __restrict__ h, int E)
{
    int e = blockIdx.x * (blockDim.x >> 5) + (threadIdx.x >> 5);
    if (e >= E) return;
    int s = src[e];
    int d = dst[e];
    if (!needed[d]) return;
    float norm = dinv[s] * dinv[d];
    int lane = threadIdx.x & 31;
    const float* xr = xw + (size_t)s * GD_HID + lane * 8;
    float*       hr = h  + (size_t)d * GD_HID + lane * 8;
    float4 v0 = *(const float4*)(xr);
    float4 v1 = *(const float4*)(xr + 4);
    atomicAdd(hr + 0, v0.x * norm);
    atomicAdd(hr + 1, v0.y * norm);
    atomicAdd(hr + 2, v0.z * norm);
    atomicAdd(hr + 3, v0.w * norm);
    atomicAdd(hr + 4, v1.x * norm);
    atomicAdd(hr + 5, v1.y * norm);
    atomicAdd(hr + 6, v1.z * norm);
    atomicAdd(hr + 7, v1.w * norm);
}

// hg[a,:] = relu(h[agent_idx[a],:])
__global__ void gather_relu(const float* __restrict__ h,
                            const int*   __restrict__ idx,
                            float* __restrict__ hg)
{
    int a = blockIdx.x;
    int c = threadIdx.x;
    float v = h[(size_t)idx[a] * GD_HID + c];
    hg[(size_t)a * GD_HID + c] = fmaxf(v, 0.0f);
}

// ---------------------------------------------------------------------------
// In-place row LayerNorm (+ optional relu). One wave per row, shfl reduction.
// ---------------------------------------------------------------------------
__global__ void ln_kernel(float* __restrict__ X,
                          const float* __restrict__ g,
                          const float* __restrict__ b,
                          int D, int relu, int rows)
{
    int row  = blockIdx.x * (blockDim.x >> 5) + (threadIdx.x >> 5);
    int lane = threadIdx.x & 31;
    if (row >= rows) return;
    float* xr = X + (size_t)row * D;

    float sum = 0.0f, sumsq = 0.0f;
    for (int c = lane; c < D; c += 32) {
        float v = xr[c];
        sum   += v;
        sumsq += v * v;
    }
#pragma unroll
    for (int off = 16; off; off >>= 1) {
        sum   += __shfl_xor(sum,   off, 32);
        sumsq += __shfl_xor(sumsq, off, 32);
    }
    float mu   = sum / (float)D;
    float var  = sumsq / (float)D - mu * mu;
    float rstd = rsqrtf(var + G_LN_EPS);

    for (int c = lane; c < D; c += 32) {
        float v = (xr[c] - mu) * rstd * g[c] + b[c];
        if (relu) v = fmaxf(v, 0.0f);
        xr[c] = v;
    }
}

// out[a] = sum_c relu(sv[a,c] + av[a,c]) * Wq[c] + bq  — wave per agent
__global__ void head_kernel(const float* __restrict__ sv,
                            const float* __restrict__ av,
                            const float* __restrict__ Wq,
                            const float* __restrict__ bq,
                            float* __restrict__ out, int nA)
{
    int a    = blockIdx.x * (blockDim.x >> 5) + (threadIdx.x >> 5);
    int lane = threadIdx.x & 31;
    if (a >= nA) return;
    float s = 0.0f;
    const float* svr = sv + (size_t)a * GFC2;
    const float* avr = av + (size_t)a * GFC2;
#pragma unroll
    for (int c = lane; c < GFC2; c += 32) {
        float v = fmaxf(svr[c] + avr[c], 0.0f);
        s += v * Wq[c];
    }
#pragma unroll
    for (int off = 16; off; off >>= 1) s += __shfl_xor(s, off, 32);
    if (lane == 0) out[a] = s + bq[0];
}

// ---------------------------------------------------------------------------
// Launcher
// ---------------------------------------------------------------------------
extern "C" void kernel_launch(void* const* d_in, const int* in_sizes, int n_in,
                              void* d_out, int out_size, void* d_ws, size_t ws_size,
                              hipStream_t stream)
{
    const float* x    = (const float*)d_in[0];
    const int*   ei   = (const int*)  d_in[1];   // [2, E] : src row, dst row
    const float* act  = (const float*)d_in[2];
    const int*   aidx = (const int*)  d_in[3];
    const float* Wg   = (const float*)d_in[4];
    const float* bg   = (const float*)d_in[5];
    const float* W1   = (const float*)d_in[6];
    const float* b1   = (const float*)d_in[7];
    const float* g1   = (const float*)d_in[8];
    const float* be1  = (const float*)d_in[9];
    const float* W2   = (const float*)d_in[10];
    const float* b2   = (const float*)d_in[11];
    const float* g2   = (const float*)d_in[12];
    const float* be2  = (const float*)d_in[13];
    const float* Wa   = (const float*)d_in[14];
    const float* ba   = (const float*)d_in[15];
    const float* Wq   = (const float*)d_in[16];
    const float* bq   = (const float*)d_in[17];
    float* out = (float*)d_out;

    const int N = GN_NODES;
    const int E = in_sizes[1] / 2;
    const int A = GN_AGENTS;

    // ---- scratch carve-out ----
    char*  ws  = (char*)d_ws;
    size_t off = 0;
    auto carve = [&](size_t bytes) -> void* {
        void* p = ws + off;
        off = (off + bytes + 255) & ~(size_t)255;
        return p;
    };
    float*  deg    = (float*) carve((size_t)N * 4);           // -> dinv in place
    int*    needed = (int*)   carve((size_t)N * 4);
    float*  xw     = (float*) carve((size_t)N * GD_HID * 4);
    float*  h      = (float*) carve((size_t)N * GD_HID * 4);
    float*  hg     = (float*) carve((size_t)A * GD_HID * 4);
    float*  t1     = (float*) carve((size_t)A * GFC1  * 4);
    float*  t2     = (float*) carve((size_t)A * GFC2  * 4);
    float*  t3     = (float*) carve((size_t)A * GFC2  * 4);
    __bf16* pWg    = (__bf16*)carve((size_t)GD_IN  * GD_HID * 2);
    __bf16* pW1    = (__bf16*)carve((size_t)GD_HID * GFC1  * 2);
    __bf16* pW2    = (__bf16*)carve((size_t)GFC1   * GFC2  * 2);
    __bf16* pWa    = (__bf16*)carve((size_t)GN_ACT * GFC2  * 2);
    (void)ws_size; (void)n_in; (void)out_size;

    const int* srcp = ei;
    const int* dstp = ei + E;

    // ---- 0. pack weight matrices into WMMA bf16 fragment order ----
    {
        int tg = (GD_IN / 32) * (GD_HID / 16);
        pack_b<<<(tg + 7) / 8, 256, 0, stream>>>(Wg, pWg, GD_HID, tg);
        int t1g = (GD_HID / 32) * (GFC1 / 16);
        pack_b<<<(t1g + 7) / 8, 256, 0, stream>>>(W1, pW1, GFC1, t1g);
        int t2g = (GFC1 / 32) * (GFC2 / 16);
        pack_b<<<(t2g + 7) / 8, 256, 0, stream>>>(W2, pW2, GFC2, t2g);
        int tag = (GN_ACT / 32) * (GFC2 / 16);
        pack_b<<<(tag + 7) / 8, 256, 0, stream>>>(Wa, pWa, GFC2, tag);
    }

    // ---- 1. degrees / norms / needed mask ----
    gcn_init   <<<(N + 255) / 256, 256, 0, stream>>>(deg, needed, N);
    gcn_degree <<<(E + 255) / 256, 256, 0, stream>>>(dstp, deg, E);
    gcn_dinv   <<<(N + 255) / 256, 256, 0, stream>>>(deg, N);
    mark_needed<<<(A + 255) / 256, 256, 0, stream>>>(aidx, needed, A);

    // ---- 2. xw = x @ Wg  (50000x128 @ 128x256) ----
    gemm_bf16_wmma<<<dim3(N / 16, GD_HID / 256), 128, 0, stream>>>(
        x, pWg, nullptr, xw, GD_IN, GD_HID / 16, GD_IN, GD_HID, 0);

    // ---- 3. GCN aggregation (needed rows only) ----
    gcn_selfloop<<<N, GD_HID, 0, stream>>>(xw, deg, needed, bg, h);
    gcn_scatter <<<(E + 7) / 8, 256, 0, stream>>>(srcp, dstp, xw, deg, needed, h, E);

    // ---- 4. gather + relu ----
    gather_relu<<<A, GD_HID, 0, stream>>>(h, aidx, hg);

    // ---- 5. fc1 + LN + relu ----
    gemm_bf16_wmma<<<dim3(A / 16, GFC1 / 256), 128, 0, stream>>>(
        hg, pW1, b1, t1, GD_HID, GFC1 / 16, GD_HID, GFC1, 0);
    ln_kernel<<<(A + 7) / 8, 256, 0, stream>>>(t1, g1, be1, GFC1, 1, A);

    // ---- 6. fc2 + LN ----
    gemm_bf16_wmma<<<dim3(A / 16, GFC2 / 256), 128, 0, stream>>>(
        t1, pW2, b2, t2, GFC1, GFC2 / 16, GFC1, GFC2, 0);
    ln_kernel<<<(A + 7) / 8, 256, 0, stream>>>(t2, g2, be2, GFC2, 0, A);

    // ---- 7. action value: t3 = action @ Wa + ba ----
    gemm_bf16_wmma<<<dim3(A / 16, GFC2 / 256), 128, 0, stream>>>(
        act, pWa, ba, t3, GN_ACT, GFC2 / 16, GN_ACT, GFC2, 0);

    // ---- 8. head: out = relu(t2 + t3) @ Wq + bq ----
    head_kernel<<<(A + 7) / 8, 256, 0, stream>>>(t2, t3, Wq, bq, out, A);
}
